// PFNet7_16767552323985
// MI455X (gfx1250) — matrix operand order repaired and checked by reading, not compiled, not run on hardware
//
#include <hip/hip_runtime.h>
#include <hip/hip_bf16.h>

// ---------- vector types matching CDNA5 WMMA operand shapes ----------
typedef _Float16 h16 __attribute__((ext_vector_type(16)));  // f16 A/B frag (16x16x32)
typedef _Float16 h8v __attribute__((ext_vector_type(8)));
typedef float    v8f __attribute__((ext_vector_type(8)));   // f32 C/D frag
typedef float    v2f __attribute__((ext_vector_type(2)));   // f32 A/B frag (16x16x4)
typedef float    v4f __attribute__((ext_vector_type(4)));

#define NEG_SLOPE 0.01f

// Load a 16x16x32 f16 A-fragment for this lane from a row-major f16 buffer.
// Per ISA: lanes 0-15 hold row M=lane, K = k0+{0..7, 16..23};
//          lanes 16-31 hold row M=lane-16, K = k0+{8..15, 24..31}.
// Caller passes p = &A[row*lda + k0 + ((lane>=16)?8:0)].
__device__ __forceinline__ h16 load_a_frag(const _Float16* p) {
    h8v lo = *(const h8v*)p;
    h8v hi = *(const h8v*)(p + 16);
    h16 a;
#pragma unroll
    for (int t = 0; t < 8; ++t) { a[t] = lo[t]; a[t + 8] = hi[t]; }
    return a;
}

// ---------------------------------------------------------------------
// Generic fused GEMM:  out = act( A[N,K] @ W[wK,wN] + b )
// NCH = K/32 chunks, NT = ceil(wN/16) column tiles.
// One wave computes a 16-row tile via v_wmma_f32_16x16x32_f16.
// Weights staged in LDS pre-swizzled into B-frag order:
//   lane l holds column n = 16*nt + (l&15), K rows k0 + ((l>=16)?8:0) + {0..7, 16..23}.
// Epilogue: bias, optional LeakyReLU, up to two f16 dests (for concat
// buffers), one f32 dest with optional residual add.
// ---------------------------------------------------------------------
template <int NCH, int NT>
__global__ __launch_bounds__(256) void gemm16_kernel(
    const _Float16* __restrict__ A, int lda,
    const float* __restrict__ W, int wK, int wN,
    const float* __restrict__ bias, int relu,
    _Float16* __restrict__ outA, int ldA, int offA,
    _Float16* __restrict__ outB, int ldB, int offB,
    float* __restrict__ outF, int ldF, int offF,
    const float* __restrict__ resid, int ldR, int offR) {
    __shared__ h16 wfrag[NCH * NT][32];

    // ---- stage weights (f32 -> f16, padded with zeros) in B-frag order ----
    {
        _Float16* wf = (_Float16*)wfrag;
        const int total = NCH * NT * 512;  // frags * 32 lanes * 16 halves
        for (int idx = threadIdx.x; idx < total; idx += blockDim.x) {
            int frag = idx >> 9;
            int ln   = (idx >> 4) & 31;
            int t    = idx & 15;
            int c    = frag % NCH;
            int nt   = frag / NCH;
            int k    = 32 * c + ((ln >= 16) ? 8 : 0) + (t & 7) + ((t & 8) << 1);
            int n    = 16 * nt + (ln & 15);
            float v  = (k < wK && n < wN) ? W[(size_t)k * wN + n] : 0.f;
            wf[idx]  = (_Float16)v;
        }
    }
    __syncthreads();

    const int lane = threadIdx.x & 31;
    const int wid  = threadIdx.x >> 5;
    const int l16  = lane & 15;
    const int hiK  = (lane >= 16) ? 8 : 0;  // A-frag K offset
    const int hiM  = (lane >= 16) ? 8 : 0;  // D-frag row offset
    const int rowbase = (blockIdx.x * (blockDim.x >> 5) + wid) * 16;
    const _Float16* arow = A + (size_t)(rowbase + l16) * lda + hiK;

#pragma unroll
    for (int nt = 0; nt < NT; ++nt) {
        v8f acc = {};
#pragma unroll
        for (int c = 0; c < NCH; ++c) {
            h16 a = load_a_frag(arow + 32 * c);
            h16 b = wfrag[nt * NCH + c][lane];
            acc = __builtin_amdgcn_wmma_f32_16x16x32_f16(
                false, a, false, b, (short)0, acc, false, false);
        }
        const int coln = 16 * nt + l16;
        const float bv = (coln < wN) ? bias[coln] : 0.f;
#pragma unroll
        for (int r = 0; r < 8; ++r) {
            float v = acc[r] + bv;
            if (relu) v = (v > 0.f) ? v : NEG_SLOPE * v;
            const int row = rowbase + r + hiM;
            if (coln < wN) {
                if (outA) outA[(size_t)row * ldA + offA + coln] = (_Float16)v;
                if (outB) outB[(size_t)row * ldB + offB + coln] = (_Float16)v;
                if (outF) {
                    float o = v;
                    if (resid) o += resid[(size_t)row * ldR + offR + coln];
                    outF[(size_t)row * ldF + offF + coln] = o;
                }
            }
        }
    }
}

// ---------------- x -> f16, padded 15 -> 32 cols ----------------
__global__ __launch_bounds__(256) void prep_x_kernel(
    const float* __restrict__ x, _Float16* __restrict__ xpad, int N) {
    int idx = blockIdx.x * blockDim.x + threadIdx.x;
    if (idx >= N * 32) return;
    int row = idx >> 5, col = idx & 31;
    xpad[idx] = (col < 15) ? (_Float16)x[(size_t)row * 15 + col] : (_Float16)0.f;
}

// --------------- sn = |s|^2 ; s2 = -2*s (A prescale) ---------------
__global__ __launch_bounds__(256) void snprep_kernel(
    const float* __restrict__ s, float* __restrict__ sn,
    float* __restrict__ s2, int N) {
    int i = blockIdx.x * blockDim.x + threadIdx.x;
    if (i >= N) return;
    float a = s[(size_t)i * 4 + 0], b = s[(size_t)i * 4 + 1];
    float c = s[(size_t)i * 4 + 2], d = s[(size_t)i * 4 + 3];
    sn[i] = a * a + b * b + c * c + d * d;
    s2[(size_t)i * 4 + 0] = -2.f * a;
    s2[(size_t)i * 4 + 1] = -2.f * b;
    s2[(size_t)i * 4 + 2] = -2.f * c;
    s2[(size_t)i * 4 + 3] = -2.f * d;
}

// ---------------------------------------------------------------------
// kNN: per wave 16 queries (queries on the N axis of the WMMA).
// D = (-2*s_cand) x (s_query) + sn_cand  via v_wmma_f32_16x16x4_f32,
// so D[m][n] = sn_m - 2<s_m, s_n>  (ranking key; sn_query added later).
//
// Candidates are staged through LDS in 1024-row chunks (16 KB s2 + 4 KB sn),
// register-double-buffered: while the 8 waves run 64 WMMA tiles out of LDS,
// the next chunk's global loads are already in flight. This replaces the
// per-tile global-load -> s_wait_loadcnt 0 -> wmma latency chain with
// low-latency ds reads and cuts global candidate traffic 8x per block.
//
// Each lane (D layout: lane = query col, VGPR r = cand row r(+8)) keeps a
// sorted top-16 of its 8-row candidate stream; lane pairs merge via shfl.
// ---------------------------------------------------------------------
#define KNN_CHUNK 1024

__global__ __launch_bounds__(256) void knn_kernel(
    const float* __restrict__ s, const float* __restrict__ s2,
    const float* __restrict__ sn,
    int* __restrict__ knn_j, float* __restrict__ knn_w, int N) {
    __shared__ __align__(16) float ls2[KNN_CHUNK * 4];  // 16 KB: -2*s rows
    __shared__ __align__(16) float lsn[KNN_CHUNK];      //  4 KB: |s|^2

    const int tid  = threadIdx.x;
    const int lane = tid & 31;
    const int wid  = tid >> 5;
    const int l16  = lane & 15;
    const int hi8  = (lane >= 16) ? 8 : 0;
    const int k01  = (lane >= 16) ? 2 : 0;  // K comps {0,1} / {2,3}
    const int qbase = (blockIdx.x * (blockDim.x >> 5) + wid) * 16;
    const int q = qbase + l16;

    v2f bfrag;  // loop-invariant query fragment
    bfrag[0] = s[(size_t)q * 4 + k01];
    bfrag[1] = s[(size_t)q * 4 + k01 + 1];
    const float snq = sn[q];
#if !__has_builtin(__builtin_amdgcn_wmma_f32_16x16x4_f32)
    const float q0 = s[(size_t)q * 4 + 0], q1 = s[(size_t)q * 4 + 1];
    const float q2 = s[(size_t)q * 4 + 2], q3 = s[(size_t)q * 4 + 3];
#endif

    float d[16]; int j[16];
#pragma unroll
    for (int t = 0; t < 16; ++t) { d[t] = 3.0e38f; j[t] = 0; }

    // --- register prefetch buffers: 256 threads x (4x float4 + 4x float) ---
    const v4f* s2v = (const v4f*)s2;
    v4f r2[4]; float rn[4];
#pragma unroll
    for (int u = 0; u < 4; ++u) {  // chunk 0 prologue
        int row = u * 256 + tid;
        r2[u] = s2v[row];
        rn[u] = sn[row];
    }

    const int nchunks = N / KNN_CHUNK;
    for (int ch = 0; ch < nchunks; ++ch) {
        __syncthreads();  // previous chunk's compute finished
        {
            v4f* l2v = (v4f*)ls2;
#pragma unroll
            for (int u = 0; u < 4; ++u) {
                int row = u * 256 + tid;
                l2v[row] = r2[u];
                lsn[row] = rn[u];
            }
        }
        __syncthreads();
        if (ch + 1 < nchunks) {  // issue next chunk's loads; overlap compute
            int base = (ch + 1) * KNN_CHUNK;
#pragma unroll
            for (int u = 0; u < 4; ++u) {
                int row = u * 256 + tid;
                r2[u] = s2v[base + row];
                rn[u] = sn[base + row];
            }
        }
        const int cbase = ch * KNN_CHUNK;
        for (int t0 = 0; t0 < KNN_CHUNK; t0 += 16) {
            v2f afrag;  // candidate rows, prescaled by -2 (ds_load_b64)
            afrag[0] = ls2[(t0 + l16) * 4 + k01];
            afrag[1] = ls2[(t0 + l16) * 4 + k01 + 1];
            v8f c;
#pragma unroll
            for (int r = 0; r < 8; ++r) c[r] = lsn[t0 + hi8 + r];
            v8f dres;
#if __has_builtin(__builtin_amdgcn_wmma_f32_16x16x4_f32)
            dres = __builtin_amdgcn_wmma_f32_16x16x4_f32(
                false, afrag, false, bfrag, (short)0, c, false, false);
#else
            (void)afrag; (void)bfrag;
#pragma unroll
            for (int r = 0; r < 8; ++r) {
                const float* cs = ls2 + (t0 + hi8 + r) * 4;
                dres[r] = c[r] + 0.5f * (cs[0] * (-2.f * q0) + cs[1] * (-2.f * q1) +
                                         cs[2] * (-2.f * q2) + cs[3] * (-2.f * q3));
            }
#endif
#pragma unroll
            for (int r = 0; r < 8; ++r) {
                float key = dres[r];
                if (key < d[15]) {  // rare path: bubble-insert (static indices)
                    float kd = key; int kj = cbase + t0 + hi8 + r;
#pragma unroll
                    for (int t = 0; t < 16; ++t) {
                        bool sw = kd < d[t];
                        float td = d[t]; int tj = j[t];
                        d[t] = sw ? kd : td; j[t] = sw ? kj : tj;
                        kd   = sw ? td : kd; kj   = sw ? tj : kj;
                    }
                }
            }
        }
    }

    // merge the two disjoint sorted halves (lane l <-> lane l^16)
    float od[16]; int oj[16];
#pragma unroll
    for (int t = 0; t < 16; ++t) {
        od[t] = __shfl(d[t], lane ^ 16, 32);
        oj[t] = __shfl(j[t], lane ^ 16, 32);
    }
    if (lane < 16) {
#pragma unroll
        for (int t = 0; t < 16; ++t) {  // Batcher: top-16 = min(a[t], b[15-t])
            float a = d[t], b = od[15 - t];
            bool ta = a <= b;
            float fd = ta ? a : b;
            int   fj = ta ? j[t] : oj[15 - t];
            float w = __expf(-10.f * (fd + snq));  // exp(-10*d2)
            knn_w[(size_t)q * 16 + t] = w;
            knn_j[(size_t)q * 16 + t] = fj;
        }
    }
}

// ------------- gather + weighted mean/max -> cat192[:, 0:128) -------------
__global__ __launch_bounds__(256) void aggregate_kernel(
    const float* __restrict__ h, const int* __restrict__ knn_j,
    const float* __restrict__ knn_w, _Float16* __restrict__ cat192, int N) {
    const int lane = threadIdx.x & 31;
    const int wid  = threadIdx.x >> 5;
    const int i = blockIdx.x * (blockDim.x >> 5) + wid;
    if (i >= N) return;
    float s0 = 0.f, s1 = 0.f, m0 = -3.0e38f, m1 = -3.0e38f;
#pragma unroll
    for (int k = 0; k < 16; ++k) {
        int   jj = knn_j[(size_t)i * 16 + k];
        float w  = knn_w[(size_t)i * 16 + k];
        float v0 = w * h[(size_t)jj * 64 + lane];
        float v1 = w * h[(size_t)jj * 64 + lane + 32];
        s0 += v0; s1 += v1;
        m0 = fmaxf(m0, v0); m1 = fmaxf(m1, v1);
    }
    _Float16* o = cat192 + (size_t)i * 192;
    o[lane]       = (_Float16)(s0 * (1.f / 16.f));
    o[lane + 32]  = (_Float16)(s1 * (1.f / 16.f));
    o[64 + lane]  = (_Float16)m0;
    o[96 + lane]  = (_Float16)m1;
}

// ---------------------------------------------------------------------
extern "C" void kernel_launch(void* const* d_in, const int* in_sizes, int n_in,
                              void* d_out, int out_size, void* d_ws, size_t ws_size,
                              hipStream_t stream) {
    (void)n_in; (void)out_size; (void)ws_size;
    const float* x = (const float*)d_in[0];
    const int N = in_sizes[0] / 15;

    // params in setup_inputs() dict order: nn1(w,b x4), lin_s, lin_h, lin_out, nn2 x4, nn3 x4
    const float* w_nn1[4] = {(const float*)d_in[1], (const float*)d_in[3],
                             (const float*)d_in[5], (const float*)d_in[7]};
    const float* b_nn1[4] = {(const float*)d_in[2], (const float*)d_in[4],
                             (const float*)d_in[6], (const float*)d_in[8]};
    const float* w_lin_s = (const float*)d_in[9],  *b_lin_s = (const float*)d_in[10];
    const float* w_lin_h = (const float*)d_in[11], *b_lin_h = (const float*)d_in[12];
    const float* w_lin_o = (const float*)d_in[13], *b_lin_o = (const float*)d_in[14];
    const float* w_nn2[4] = {(const float*)d_in[15], (const float*)d_in[17],
                             (const float*)d_in[19], (const float*)d_in[21]};
    const float* b_nn2[4] = {(const float*)d_in[16], (const float*)d_in[18],
                             (const float*)d_in[20], (const float*)d_in[22]};
    const float* w_nn3[4] = {(const float*)d_in[23], (const float*)d_in[25],
                             (const float*)d_in[27], (const float*)d_in[29]};
    const float* b_nn3[4] = {(const float*)d_in[24], (const float*)d_in[26],
                             (const float*)d_in[28], (const float*)d_in[30]};

    // ---- carve workspace ----
    char* p = (char*)d_ws;
    auto carve = [&](size_t bytes) -> void* {
        void* r = (void*)p;
        p += (bytes + 255) & ~(size_t)255;
        return r;
    };
    _Float16* xpad   = (_Float16*)carve((size_t)N * 32 * 2);
    _Float16* tA     = (_Float16*)carve((size_t)N * 64 * 2);
    _Float16* tB     = (_Float16*)carve((size_t)N * 64 * 2);
    _Float16* zbuf   = (_Float16*)carve((size_t)N * 64 * 2);
    _Float16* cat192 = (_Float16*)carve((size_t)N * 192 * 2); // [agg_mean|agg_max|z]
    _Float16* catn3  = (_Float16*)carve((size_t)N * 96 * 2);  // [z2|ids|zero-pad]
    _Float16* z2     = (_Float16*)carve((size_t)N * 64 * 2);
    float* hbuf  = (float*)carve((size_t)N * 64 * 4);
    float* sbuf  = (float*)carve((size_t)N * 4 * 4);
    float* s2buf = (float*)carve((size_t)N * 4 * 4);
    float* snbuf = (float*)carve((size_t)N * 4);
    float* knnw  = (float*)carve((size_t)N * 16 * 4);
    int*   knnj  = (int*)carve((size_t)N * 16 * 4);

    float* out_ids = (float*)d_out;
    float* out_p4  = out_ids + (size_t)N * 6;

    hipMemsetAsync(catn3, 0, (size_t)N * 96 * 2, stream);  // zero pad cols 70..95

    dim3 blk(256);
    const int gG = N / 128;  // gemm grid: 8 waves * 16 rows per block

    prep_x_kernel<<<dim3((N * 32) / 256), blk, 0, stream>>>(x, xpad, N);

    // nn1 encoder: 15->64->64->64->64, LeakyReLU each
    gemm16_kernel<1, 4><<<gG, blk, 0, stream>>>(xpad, 32, w_nn1[0], 15, 64, b_nn1[0], 1,
        tA, 64, 0, nullptr, 0, 0, nullptr, 0, 0, nullptr, 0, 0);
    gemm16_kernel<2, 4><<<gG, blk, 0, stream>>>(tA, 64, w_nn1[1], 64, 64, b_nn1[1], 1,
        tB, 64, 0, nullptr, 0, 0, nullptr, 0, 0, nullptr, 0, 0);
    gemm16_kernel<2, 4><<<gG, blk, 0, stream>>>(tB, 64, w_nn1[2], 64, 64, b_nn1[2], 1,
        tA, 64, 0, nullptr, 0, 0, nullptr, 0, 0, nullptr, 0, 0);
    gemm16_kernel<2, 4><<<gG, blk, 0, stream>>>(tA, 64, w_nn1[3], 64, 64, b_nn1[3], 1,
        zbuf, 64, 0, cat192, 192, 128, nullptr, 0, 0, nullptr, 0, 0); // z also -> cat192[:,128:192)

    // GravNet projections
    gemm16_kernel<2, 4><<<gG, blk, 0, stream>>>(zbuf, 64, w_lin_h, 64, 64, b_lin_h, 0,
        nullptr, 0, 0, nullptr, 0, 0, hbuf, 64, 0, nullptr, 0, 0);     // h (f32)
    gemm16_kernel<2, 1><<<gG, blk, 0, stream>>>(zbuf, 64, w_lin_s, 64, 4, b_lin_s, 0,
        nullptr, 0, 0, nullptr, 0, 0, sbuf, 4, 0, nullptr, 0, 0);      // s (f32)

    snprep_kernel<<<dim3((N + 255) / 256), blk, 0, stream>>>(sbuf, snbuf, s2buf, N);
    knn_kernel<<<dim3(N / 128), blk, 0, stream>>>(sbuf, s2buf, snbuf, knnj, knnw, N);
    aggregate_kernel<<<dim3(N / 8), blk, 0, stream>>>(hbuf, knnj, knnw, cat192, N);

    // lin_out on [agg_mean|agg_max|z] (K=192) + LeakyReLU -> z2, also -> catn3[:,0:64)
    gemm16_kernel<6, 4><<<gG, blk, 0, stream>>>(cat192, 192, w_lin_o, 192, 64, b_lin_o, 1,
        z2, 64, 0, catn3, 96, 0, nullptr, 0, 0, nullptr, 0, 0);

    // nn2: 3x (64->64 + LReLU), then 64->6 plain -> d_out ids, f16 copy -> catn3[:,64:70)
    gemm16_kernel<2, 4><<<gG, blk, 0, stream>>>(z2, 64, w_nn2[0], 64, 64, b_nn2[0], 1,
        tA, 64, 0, nullptr, 0, 0, nullptr, 0, 0, nullptr, 0, 0);
    gemm16_kernel<2, 4><<<gG, blk, 0, stream>>>(tA, 64, w_nn2[1], 64, 64, b_nn2[1], 1,
        tB, 64, 0, nullptr, 0, 0, nullptr, 0, 0, nullptr, 0, 0);
    gemm16_kernel<2, 4><<<gG, blk, 0, stream>>>(tB, 64, w_nn2[2], 64, 64, b_nn2[2], 1,
        tA, 64, 0, nullptr, 0, 0, nullptr, 0, 0, nullptr, 0, 0);
    gemm16_kernel<2, 1><<<gG, blk, 0, stream>>>(tA, 64, w_nn2[3], 64, 6, b_nn2[3], 0,
        catn3, 96, 64, nullptr, 0, 0, out_ids, 6, 0, nullptr, 0, 0);

    // nn3 on [z2|ids] (K=70 padded to 96): 3x LReLU layers, then 64->4 + residual x[:,3:7)
    gemm16_kernel<3, 4><<<gG, blk, 0, stream>>>(catn3, 96, w_nn3[0], 70, 64, b_nn3[0], 1,
        tA, 64, 0, nullptr, 0, 0, nullptr, 0, 0, nullptr, 0, 0);
    gemm16_kernel<2, 4><<<gG, blk, 0, stream>>>(tA, 64, w_nn3[1], 64, 64, b_nn3[1], 1,
        tB, 64, 0, nullptr, 0, 0, nullptr, 0, 0, nullptr, 0, 0);
    gemm16_kernel<2, 4><<<gG, blk, 0, stream>>>(tB, 64, w_nn3[2], 64, 64, b_nn3[2], 1,
        tA, 64, 0, nullptr, 0, 0, nullptr, 0, 0, nullptr, 0, 0);
    gemm16_kernel<2, 1><<<gG, blk, 0, stream>>>(tA, 64, w_nn3[3], 64, 4, b_nn3[3], 0,
        nullptr, 0, 0, nullptr, 0, 0, out_p4, 4, 0, x, 15, 3);
}
